// OlMoEAttention_81097572483288
// MI455X (gfx1250) — compile-verified, matrix operand order checked by reading, not run on hardware
//
#include <hip/hip_runtime.h>
#include <math.h>

// ---------------------------------------------------------------------------
// OlMoE attention for MI455X (gfx1250): bf16 WMMA + TDM async tile staging.
// Pipeline: cvt->GEMM(QKV)->RMSNorm+RoPE->flash-attn(WMMA)->GEMM(out proj)
// ---------------------------------------------------------------------------

typedef __bf16 bf16_t;
typedef __attribute__((ext_vector_type(16))) __bf16 v16bf;
typedef __attribute__((ext_vector_type(8)))  __bf16 v8bf;
typedef __attribute__((ext_vector_type(8)))  float  v8f;

#define S_LEN 4096
#define HDIM  2048
#define NHEAD 16
#define NKVH  4
#define HD    128
#define NREP  4
#define EPSV  1e-5f
#define LOG_THETA 9.210340371976184f   // ln(10000)

// ---------------------------------------------------------------------------
// Tensor Data Mover staging (preferred) / async-to-LDS fallback
// ---------------------------------------------------------------------------
#if defined(__has_builtin)
#if __has_builtin(__builtin_amdgcn_tensor_load_to_lds) && \
    __has_builtin(__builtin_amdgcn_s_wait_tensorcnt)
#define USE_TDM 1
#endif
#endif
#ifndef USE_TDM
#define USE_TDM 0
#endif

#if USE_TDM
typedef unsigned int tdm_v4u __attribute__((ext_vector_type(4)));
typedef int          tdm_v8i __attribute__((ext_vector_type(8)));
typedef int          tdm_v4i __attribute__((ext_vector_type(4)));

// 2D tile load: tile_d1 rows of tile_d0 bf16 elements, row stride
// `stride_elems` in global memory, packed contiguously into LDS at lds_off.
// D# packing per CDNA5 ISA ch.8 (group0: count/lds/global/type, group1:
// data_size, tensor dims (== tile dims; we never read OOB), tile dims, stride).
// clang-23 / therock-10.0 arity: (v4u, v8i, v4i, v4i, v8i, i32 cpol)
static __device__ inline void tdm_load_2d(unsigned int lds_off, const void* gptr,
                                          unsigned int tile_d0, unsigned int tile_d1,
                                          unsigned long long stride_elems) {
  unsigned long long ga = (unsigned long long)(size_t)gptr;
  tdm_v4u g0;
  g0[0] = 1u;                                          // count=1, no gather
  g0[1] = lds_off;                                     // lds_addr (bytes)
  g0[2] = (unsigned int)ga;                            // global_addr[31:0]
  g0[3] = (unsigned int)((ga >> 32) & 0x1FFFFFFull) |  // global_addr[56:32]
          (2u << 30);                                  // type = 2 ("image")
  unsigned int td0 = tile_d0;
  unsigned int td1 = (tile_d1 == 0) ? 1u : tile_d1;
  tdm_v8i g1;
  g1[0] = (int)(1u << 16);                             // data_size=1 (2 bytes)
  g1[1] = (int)((td0 & 0xFFFFu) << 16);                // tensor_dim0[15:0]
  g1[2] = (int)(((td0 >> 16) & 0xFFFFu) |              // tensor_dim0[31:16]
                ((td1 & 0xFFFFu) << 16));              // tensor_dim1[15:0]
  g1[3] = (int)(((td1 >> 16) & 0xFFFFu) |              // tensor_dim1[31:16]
                ((tile_d0 & 0xFFFFu) << 16));          // tile_dim0
  g1[4] = (int)(tile_d1 & 0xFFFFu);                    // tile_dim1 (tile_dim2=0)
  g1[5] = (int)(unsigned int)(stride_elems & 0xFFFFFFFFull);   // dim0_stride lo
  g1[6] = (int)(unsigned int)((stride_elems >> 32) & 0xFFFFull); // dim0_stride hi
  g1[7] = 0;
  tdm_v4i z4 = {0, 0, 0, 0};
  tdm_v8i z8 = {0, 0, 0, 0, 0, 0, 0, 0};
  __builtin_amdgcn_tensor_load_to_lds(g0, g1, z4, z4, z8, 0);
}
#define TDM_WAIT() __builtin_amdgcn_s_wait_tensorcnt(0)
#else
// Per-lane 16B async copy global->LDS (ASYNCcnt), GV addressing mode.
static __device__ inline void async_ld16(unsigned int lds_off, const void* g) {
  asm volatile("global_load_async_to_lds_b128 %0, %1, off"
               :: "v"(lds_off), "v"((unsigned long long)(size_t)g) : "memory");
}
static __device__ inline void async_wait0() {
  asm volatile("s_wait_asynccnt 0" ::: "memory");
}
#endif

// ---------------------------------------------------------------------------
// Fragment loader: 16x32 bf16 A/B fragment from an LDS tile with row stride
// STRIDE. Per CDNA5 ISA 7.12.2: lane L (L<16) holds row L, K = {kb..kb+7,
// kb+16..kb+23} with kb = (lane>=16)*8, i.e. two 16-byte chunks per lane.
// ---------------------------------------------------------------------------
template<int STRIDE>
static __device__ inline v16bf load_frag(const bf16_t* base, int row, int koff) {
  const bf16_t* p = base + row * STRIDE + koff;
  v8bf lo = *reinterpret_cast<const v8bf*>(p);
  v8bf hi = *reinterpret_cast<const v8bf*>(p + 16);
  v16bf f;
#pragma unroll
  for (int i = 0; i < 8; ++i) { f[i] = lo[i]; f[i + 8] = hi[i]; }
  return f;
}

// ---------------------------------------------------------------------------
// fp32 -> bf16 conversion (n multiple of 1024)
// ---------------------------------------------------------------------------
__global__ __launch_bounds__(256)
void cvt_f32_bf16(const float* __restrict__ in, bf16_t* __restrict__ out, int n) {
  int i = (blockIdx.x * 256 + threadIdx.x) * 4;
  if (i + 3 < n) {
    float4 v = *reinterpret_cast<const float4*>(in + i);
    out[i + 0] = (bf16_t)v.x;
    out[i + 1] = (bf16_t)v.y;
    out[i + 2] = (bf16_t)v.z;
    out[i + 3] = (bf16_t)v.w;
  }
}

// ---------------------------------------------------------------------------
// C[M,N] = A[M,K] * W[N,K]^T   (bf16 in, fp32 out)
// Block tile 128x64, K-step 32, 256 threads = 8 waves (4x2 of 32x32 regions).
// Tiles staged by the Tensor Data Mover (one 2D DMA per operand per K-step).
// ---------------------------------------------------------------------------
#define GTM 128
#define GTN 64
#define GTK 32

__global__ __launch_bounds__(256)
void gemm_bf16_nt(const bf16_t* __restrict__ A, const bf16_t* __restrict__ W,
                  float* __restrict__ C, int M, int N, int K) {
  __shared__ __align__(16) bf16_t As[GTM][GTK];
  __shared__ __align__(16) bf16_t Ws[GTN][GTK];

  const int tid  = threadIdx.x;
  const int wave = tid >> 5;
  const int lane = tid & 31;
  const int wrow = wave >> 1;          // 0..3  (x32 rows)
  const int wcol = wave & 1;           // 0..1  (x32 cols)
  const int lrow = lane & 15;
  const int kb   = (lane >> 4) * 8;
  const int cn   = lane & 15;
  const int mh   = (lane >> 4) * 8;
  const size_t m0 = (size_t)blockIdx.y * GTM;
  const size_t n0 = (size_t)blockIdx.x * GTN;

  v8f acc[2][2] = {};

  const int aidx = tid * 16;           // 256*16 = 4096 = 128*32
  const int ar = aidx >> 5, ac = aidx & 31;
#if !USE_TDM
  const int widx = tid * 8;            // 256*8 = 2048 = 64*32
  const int wr = widx >> 5, wc = widx & 31;
#endif

  for (int k0 = 0; k0 < K; k0 += GTK) {
#if USE_TDM
    if (wave == 0) {
      tdm_load_2d((unsigned int)(size_t)&As[0][0], &A[m0 * K + k0],
                  GTK, GTM, (unsigned long long)K);
      tdm_load_2d((unsigned int)(size_t)&Ws[0][0], &W[n0 * K + k0],
                  GTK, GTN, (unsigned long long)K);
      TDM_WAIT();
    }
#else
    async_ld16((unsigned int)(size_t)&As[ar][ac],     &A[(m0 + ar) * K + k0 + ac]);
    async_ld16((unsigned int)(size_t)&As[ar][ac + 8], &A[(m0 + ar) * K + k0 + ac + 8]);
    async_ld16((unsigned int)(size_t)&Ws[wr][wc],     &W[(n0 + wr) * K + k0 + wc]);
    async_wait0();
#endif
    __syncthreads();

    // prefetch next K-slab of A into GL2 (global_prefetch_b8)
    if (k0 + GTK < K)
      __builtin_prefetch(&A[(m0 + ar) * K + k0 + GTK + ac], 0, 1);

#pragma unroll
    for (int i = 0; i < 2; ++i) {
      v16bf a = load_frag<GTK>(&As[wrow * 32 + i * 16][0], lrow, kb);
#pragma unroll
      for (int j = 0; j < 2; ++j) {
        v16bf b = load_frag<GTK>(&Ws[wcol * 32 + j * 16][0], lrow, kb);
        acc[i][j] = __builtin_amdgcn_wmma_f32_16x16x32_bf16(
            false, a, false, b, (short)0, acc[i][j], false, false);
      }
    }
    __syncthreads();
  }

  // C/D layout: vgpr r -> M = r + 8*(lane>=16), N = lane&15
#pragma unroll
  for (int i = 0; i < 2; ++i)
#pragma unroll
    for (int j = 0; j < 2; ++j) {
      size_t mb = m0 + wrow * 32 + i * 16 + mh;
      size_t nb = n0 + wcol * 32 + j * 16 + cn;
#pragma unroll
      for (int r = 0; r < 8; ++r)
        C[(mb + r) * N + nb] = acc[i][j][r];
    }
}

// ---------------------------------------------------------------------------
// RMSNorm (over WIDTH) + RoPE, fp32 in -> bf16 out in [NHEADS][S][HD] layout
// ---------------------------------------------------------------------------
template<int WIDTH, int NHEADS>
__global__ __launch_bounds__(256)
void norm_rope(const float* __restrict__ raw, const float* __restrict__ w,
               bf16_t* __restrict__ out) {
  const int row = blockIdx.x;
  const int tid = threadIdx.x;
  const float* x = raw + (size_t)row * WIDTH;
  __shared__ float red[8];

  float ss = 0.f;
  for (int i = tid; i < WIDTH; i += 256) { float v = x[i]; ss += v * v; }
#pragma unroll
  for (int off = 16; off > 0; off >>= 1) ss += __shfl_xor(ss, off);
  if ((tid & 31) == 0) red[tid >> 5] = ss;
  __syncthreads();
  float tot = 0.f;
#pragma unroll
  for (int i = 0; i < 8; ++i) tot += red[i];
  const float rinv = rsqrtf(tot / (float)WIDTH + EPSV);

  for (int i = tid; i < WIDTH; i += 256) {
    int hh = i / HD, d = i % HD;
    int dd = d & 63;
    int partner = (d < 64) ? i + 64 : i - 64;
    float v  = x[i] * rinv * w[i];
    float pv = x[partner] * rinv * w[partner];
    float freq = __expf(-(2.0f * (float)dd / (float)HD) * LOG_THETA);
    float ang = (float)row * freq;
    float cv = __cosf(ang), sv = __sinf(ang);
    float o = (d < 64) ? (v * cv - pv * sv) : (v * cv + pv * sv);
    out[((size_t)hh * S_LEN + row) * HD + d] = (bf16_t)o;
  }
}

// ---------------------------------------------------------------------------
// V fp32 [S][NKV*HD] -> bf16 transposed [NKV][HD][S] (K-contiguous for PV WMMA)
// ---------------------------------------------------------------------------
__global__ __launch_bounds__(256)
void v_transpose(const float* __restrict__ vraw, bf16_t* __restrict__ vt) {
  size_t i = (size_t)blockIdx.x * 256 + threadIdx.x;   // over NKV*HD*S
  int s = (int)(i & (S_LEN - 1));
  size_t gd = i >> 12;                                 // S_LEN == 2^12
  vt[i] = (bf16_t)vraw[(size_t)s * (NKVH * HD) + gd];
}

// ---------------------------------------------------------------------------
// Flash attention: per (head, 64-row q tile) block, 4 waves x 16 rows.
// Q [NH][S][HD], K [NKV][S][HD], Vt [NKV][HD][S], O bf16 [S][NH*HD]
// Q/K tiles: one 1D TDM DMA (16KB contiguous); Vt tile: 2D TDM (128 x 64).
// ---------------------------------------------------------------------------
__global__ __launch_bounds__(128)
void flash_attn(const bf16_t* __restrict__ Q, const bf16_t* __restrict__ K,
                const bf16_t* __restrict__ Vt, bf16_t* __restrict__ O) {
  __shared__ __align__(16) bf16_t Qs[64][HD];
  __shared__ __align__(16) bf16_t Ks[64][HD];
  __shared__ __align__(16) bf16_t Vts[HD][64];
  __shared__ __align__(16) float  Ss[64][64];
  __shared__ __align__(16) bf16_t Ps[64][64];
  __shared__ float m_s[64], l_s[64], corr_s[64];

  const int tid  = threadIdx.x;
  const int wave = tid >> 5;
  const int lane = tid & 31;
  const int lrow = lane & 15;
  const int kb   = (lane >> 4) * 8;
  const int cn   = lane & 15;
  const int mh   = (lane >> 4) * 8;
  const int h  = blockIdx.y;
  const int g  = h / NREP;
  const int q0 = blockIdx.x * 64;
  const int qr0 = wave * 16;
  const float scale = 0.088388347648318447f;   // 1/sqrt(128)

  // stage Q tile (contiguous 16KB)
#if USE_TDM
  if (wave == 0) {
    tdm_load_2d((unsigned int)(size_t)&Qs[0][0], Q + ((size_t)h * S_LEN + q0) * HD,
                64 * HD, 1, (unsigned long long)(64 * HD));
    TDM_WAIT();
  }
#else
  {
    const bf16_t* src = Q + ((size_t)h * S_LEN + q0) * HD;
    for (int i = tid * 8; i < 64 * HD; i += 128 * 8)
      async_ld16((unsigned int)(size_t)(&Qs[0][0] + i), src + i);
  }
#endif
  if (tid < 64) { m_s[tid] = -3.0e38f; l_s[tid] = 0.f; }

  v8f acc[8] = {};

  const int nkt = blockIdx.x + 1;   // causal: only k tiles with k0 <= q0
  for (int kt = 0; kt < nkt; ++kt) {
    const int k0 = kt * 64;
#if USE_TDM
    if (wave == 0) {
      tdm_load_2d((unsigned int)(size_t)&Ks[0][0],
                  K + ((size_t)g * S_LEN + k0) * HD,
                  64 * HD, 1, (unsigned long long)(64 * HD));
      tdm_load_2d((unsigned int)(size_t)&Vts[0][0],
                  Vt + (size_t)g * HD * S_LEN + k0,
                  64, HD, (unsigned long long)S_LEN);
      TDM_WAIT();
    }
#else
    {
      const bf16_t* src = K + ((size_t)g * S_LEN + k0) * HD;
      for (int i = tid * 8; i < 64 * HD; i += 128 * 8)
        async_ld16((unsigned int)(size_t)(&Ks[0][0] + i), src + i);
      const bf16_t* vsrc = Vt + ((size_t)g * HD + tid) * S_LEN + k0;
#pragma unroll
      for (int c = 0; c < 8; ++c)
        async_ld16((unsigned int)(size_t)(&Vts[tid][0] + c * 8), vsrc + c * 8);
      async_wait0();
    }
#endif
    __syncthreads();

    // S = Q * K^T : 16 rows x 64 cols per wave, HD=128 -> 4 K-steps
    v8f sc[4] = {};
#pragma unroll
    for (int n = 0; n < 4; ++n)
#pragma unroll
      for (int ks = 0; ks < 4; ++ks) {
        v16bf a = load_frag<HD>(&Qs[qr0][0],     lrow, ks * 32 + kb);
        v16bf b = load_frag<HD>(&Ks[n * 16][0],  lrow, ks * 32 + kb);
        sc[n] = __builtin_amdgcn_wmma_f32_16x16x32_bf16(
            false, a, false, b, (short)0, sc[n], false, false);
      }

    // masked, scaled scores -> LDS
#pragma unroll
    for (int n = 0; n < 4; ++n)
#pragma unroll
      for (int r = 0; r < 8; ++r) {
        int row = qr0 + mh + r;
        int col = n * 16 + cn;
        float v = sc[n][r] * scale;
        if (q0 + row < k0 + col) v = -3.0e38f;
        Ss[row][col] = v;
      }
    __syncthreads();

    // online softmax (one thread per row)
    if (tid < 64) {
      float mprev = m_s[tid];
      float rmax = mprev;
      for (int c = 0; c < 64; ++c) rmax = fmaxf(rmax, Ss[tid][c]);
      float corr = __expf(mprev - rmax);
      float l = l_s[tid] * corr;
      for (int c = 0; c < 64; ++c) {
        float p = __expf(Ss[tid][c] - rmax);
        Ps[tid][c] = (bf16_t)p;
        l += p;
      }
      m_s[tid] = rmax; l_s[tid] = l; corr_s[tid] = corr;
    }
    __syncthreads();

    // rescale O fragments, then O += P * Vt  (P k-dim 64 -> 2 K-steps)
#pragma unroll
    for (int j = 0; j < 8; ++j)
#pragma unroll
      for (int r = 0; r < 8; ++r)
        acc[j][r] *= corr_s[qr0 + mh + r];

#pragma unroll
    for (int j = 0; j < 8; ++j)
#pragma unroll
      for (int ks = 0; ks < 2; ++ks) {
        v16bf a = load_frag<64>(&Ps[qr0][0],      lrow, ks * 32 + kb);
        v16bf b = load_frag<64>(&Vts[j * 16][0],  lrow, ks * 32 + kb);
        acc[j] = __builtin_amdgcn_wmma_f32_16x16x32_bf16(
            false, a, false, b, (short)0, acc[j], false, false);
      }
    __syncthreads();
  }

  // epilogue: O / l, write bf16 [S][NH*HD]
#pragma unroll
  for (int j = 0; j < 8; ++j)
#pragma unroll
    for (int r = 0; r < 8; ++r) {
      int row = qr0 + mh + r;
      float inv = 1.0f / l_s[row];
      O[(size_t)(q0 + row) * (NHEAD * HD) + h * HD + j * 16 + cn] =
          (bf16_t)(acc[j][r] * inv);
    }
}

// ---------------------------------------------------------------------------
// host-side launcher
// ---------------------------------------------------------------------------
extern "C" void kernel_launch(void* const* d_in, const int* in_sizes, int n_in,
                              void* d_out, int out_size, void* d_ws, size_t ws_size,
                              hipStream_t stream) {
  const float* hs = (const float*)d_in[0];
  const float* wq = (const float*)d_in[1];
  const float* wk = (const float*)d_in[2];
  const float* wv = (const float*)d_in[3];
  const float* wo = (const float*)d_in[4];
  const float* qw = (const float*)d_in[5];
  const float* kw = (const float*)d_in[6];
  float* out = (float*)d_out;

  char* ws = (char*)d_ws;
  auto take = [&](size_t bytes) { char* p = ws; ws += bytes; return p; };

  bf16_t* hs_b  = (bf16_t*)take((size_t)S_LEN * HDIM * 2);
  bf16_t* wq_b  = (bf16_t*)take((size_t)HDIM * HDIM * 2);
  bf16_t* wk_b  = (bf16_t*)take((size_t)(NKVH * HD) * HDIM * 2);
  bf16_t* wv_b  = (bf16_t*)take((size_t)(NKVH * HD) * HDIM * 2);
  bf16_t* wo_b  = (bf16_t*)take((size_t)HDIM * HDIM * 2);
  float*  q_raw = (float*)take((size_t)S_LEN * HDIM * 4);
  float*  k_raw = (float*)take((size_t)S_LEN * (NKVH * HD) * 4);
  float*  v_raw = (float*)take((size_t)S_LEN * (NKVH * HD) * 4);
  bf16_t* qh    = (bf16_t*)take((size_t)NHEAD * S_LEN * HD * 2);
  bf16_t* kh    = (bf16_t*)take((size_t)NKVH * S_LEN * HD * 2);
  bf16_t* vt    = (bf16_t*)take((size_t)NKVH * HD * S_LEN * 2);
  bf16_t* attn  = (bf16_t*)take((size_t)S_LEN * HDIM * 2);
  (void)ws_size; (void)in_sizes; (void)n_in; (void)out_size;

  // 1) fp32 -> bf16
  {
    int n;
    n = S_LEN * HDIM;        cvt_f32_bf16<<<n / 1024, 256, 0, stream>>>(hs, hs_b, n);
    n = HDIM * HDIM;         cvt_f32_bf16<<<n / 1024, 256, 0, stream>>>(wq, wq_b, n);
    n = NKVH * HD * HDIM;    cvt_f32_bf16<<<n / 1024, 256, 0, stream>>>(wk, wk_b, n);
    n = NKVH * HD * HDIM;    cvt_f32_bf16<<<n / 1024, 256, 0, stream>>>(wv, wv_b, n);
    n = HDIM * HDIM;         cvt_f32_bf16<<<n / 1024, 256, 0, stream>>>(wo, wo_b, n);
  }

  // 2) QKV projections (WMMA GEMMs)
  gemm_bf16_nt<<<dim3(HDIM / GTN, S_LEN / GTM), 256, 0, stream>>>(
      hs_b, wq_b, q_raw, S_LEN, HDIM, HDIM);
  gemm_bf16_nt<<<dim3((NKVH * HD) / GTN, S_LEN / GTM), 256, 0, stream>>>(
      hs_b, wk_b, k_raw, S_LEN, NKVH * HD, HDIM);
  gemm_bf16_nt<<<dim3((NKVH * HD) / GTN, S_LEN / GTM), 256, 0, stream>>>(
      hs_b, wv_b, v_raw, S_LEN, NKVH * HD, HDIM);

  // 3) RMSNorm + RoPE (Q,K) and V transpose
  norm_rope<HDIM, NHEAD><<<S_LEN, 256, 0, stream>>>(q_raw, qw, qh);
  norm_rope<NKVH * HD, NKVH><<<S_LEN, 256, 0, stream>>>(k_raw, kw, kh);
  v_transpose<<<(NKVH * HD * S_LEN) / 256, 256, 0, stream>>>(v_raw, vt);

  // 4) causal flash attention (WMMA)
  flash_attn<<<dim3(S_LEN / 64, NHEAD), 128, 0, stream>>>(qh, kh, vt, attn);

  // 5) output projection (WMMA GEMM, fp32 out)
  gemm_bf16_nt<<<dim3(HDIM / GTN, S_LEN / GTM), 256, 0, stream>>>(
      attn, wo_b, out, S_LEN, HDIM, HDIM);
}
